// BiLSTMCRF_74766790689477
// MI455X (gfx1250) — matrix operand order repaired and checked by reading, not compiled, not run on hardware
//
#include <hip/hip_runtime.h>
#include <hip/hip_bf16.h>

// ---------------- problem constants ----------------
#define Bb   32
#define Ss   512
#define Ee   256
#define Hh   512
#define Ll   32
#define HDd  256
#define NG   1024          // 4*HD gates
#define Mrows (Bb*Ss)      // 16384 tokens

// ---------------- WMMA types / helpers ----------------
typedef __attribute__((ext_vector_type(16))) __bf16 v16bf;
typedef __attribute__((ext_vector_type(8)))  float  v8f;

union BFragU { uint4 q[2]; v16bf v; };

// A-matrix 16x32 bf16 fragment (ISA layout):
// lane L<16 -> row L, elems 0..7 = K[kb..kb+7], elems 8..15 = K[kb+16..kb+23], kb = 0
// lane L>=16 -> row L-16, kb = 8.  Caller passes p = &A[row*ld + kb].
__device__ __forceinline__ v16bf ld_a_frag(const __bf16* p) {
    BFragU u;
    u.q[0] = *(const uint4*)(p);
    u.q[1] = *(const uint4*)(p + 16);
    return u.v;
}
// B-matrix 32x16 bf16 fragment: lane = output column; elems 0..7 = K[kb..kb+7], elems 8..15 = K[kb+16..kb+23]
// of W[col] (W stored as N x K). Caller passes p = &W[col*K + kb], kb = (lane>=16)*8.
__device__ __forceinline__ v16bf ld_b_frag(const __bf16* p) {
    BFragU u;
    u.q[0] = *(const uint4*)(p);
    u.q[1] = *(const uint4*)(p + 16);
    return u.v;
}
__device__ __forceinline__ v8f wmma_bf16(v16bf a, v16bf b, v8f c) {
    return __builtin_amdgcn_wmma_f32_16x16x32_bf16(
        /*neg_a=*/false, a, /*neg_b=*/false, b,
        /*c_mod=*/(short)0, c, /*reuse_a=*/false, /*reuse_b=*/false);
}
__device__ __forceinline__ float sigmoidf_fast(float x) {
    return 1.0f / (1.0f + __expf(-x));
}

// ---------------- elementwise prep kernels ----------------
__global__ void cvt_bf16_kernel(const float* __restrict__ src,
                                __hip_bfloat16* __restrict__ dst, int n) {
    for (int i = blockIdx.x * blockDim.x + threadIdx.x; i < n;
         i += gridDim.x * blockDim.x)
        dst[i] = __float2bfloat16(src[i]);
}

__global__ void embed_kernel(const int* __restrict__ x,
                             const float* __restrict__ emb,
                             __hip_bfloat16* __restrict__ dst) {
    const int total = Mrows * Ee;
    for (int i = blockIdx.x * blockDim.x + threadIdx.x; i < total;
         i += gridDim.x * blockDim.x) {
        int tok = x[i >> 8];        // / Ee (Ee == 256)
        int e   = i & (Ee - 1);
        dst[i]  = __float2bfloat16(emb[(size_t)tok * Ee + e]);
    }
}

// ---------------- generic WMMA GEMM: C(MxN,f32) = A(MxK,bf16) * W(NxK,bf16)^T + bias ----------------
// block = 128 threads = 4 waves; wave w computes cols [ (blockIdx.x*4+w)*16 , +16 ),
// rows [ blockIdx.y*16 , +16 ).
__global__ __launch_bounds__(128) void wmma_gemm_bias_kernel(
    const __bf16* __restrict__ A, const __bf16* __restrict__ W,
    const float* __restrict__ bias1, const float* __restrict__ bias2,
    float* __restrict__ C, int M, int N, int K) {
    const int lane    = threadIdx.x & 31;
    const int wv      = threadIdx.x >> 5;
    const int colTile = blockIdx.x * 4 + wv;
    if (colTile * 16 >= N) return;
    const int rowBase = blockIdx.y * 16;
    const int colBase = colTile * 16;
    const int hi      = lane >> 4;            // 0 or 1 (lane half)
    const int aRow    = rowBase + (lane & 15);
    const int aK      = hi * 8;
    const int bCol    = colBase + (lane & 15);
    const int bK      = hi * 8;
    const __bf16* aPtr = A + (size_t)aRow * K + aK;
    const __bf16* bPtr = W + (size_t)bCol * K + bK;

    v8f acc = {0.f, 0.f, 0.f, 0.f, 0.f, 0.f, 0.f, 0.f};
    for (int k0 = 0; k0 < K; k0 += 32) {
        if (k0 + 64 < K) __builtin_prefetch(bPtr + k0 + 64, 0, 1); // global_prefetch
        v16bf af = ld_a_frag(aPtr + k0);
        v16bf bf = ld_b_frag(bPtr + k0);
        acc = wmma_bf16(af, bf, acc);
    }
    const int col  = colBase + (lane & 15);
    float bval = (bias1 ? bias1[col] : 0.f) + (bias2 ? bias2[col] : 0.f);
    const int r0 = hi * 8;
#pragma unroll
    for (int r = 0; r < 8; ++r) {
        int row = rowBase + r0 + r;
        C[(size_t)row * N + col] = acc[r] + bval;
    }
}

// ---------------- persistent bidirectional LSTM scan ----------------
// grid.x = 2 (direction). One workgroup (8 waves) owns the whole 32x1024 recurrent
// GEMM each step; h (bf16) / c (f32) live in LDS for all 512 steps; W_hh streams
// from L2 through WMMA.
__global__ __launch_bounds__(256) void lstm_scan_kernel(
    const float* __restrict__ gxAll,    // (2, B, S, NG), biases folded in
    const __bf16* __restrict__ whhAll,  // (2, NG, HD)
    const int* __restrict__ xlen,       // (B)
    __hip_bfloat16* __restrict__ out,   // (B, S, H) pre-zeroed
    int S_) {
    const int dir = blockIdx.x;
    const float*  gx  = gxAll + (size_t)dir * Bb * S_ * NG;
    const __bf16* whh = whhAll + (size_t)dir * NG * HDd;
    const int dirOff  = dir * HDd;

    extern __shared__ char smem[];
    __bf16* h_sh = (__bf16*)smem;                               // 32*256 bf16 = 16KB
    float*  c_sh = (float*)(smem + Bb * HDd * 2);               // 32*256 f32  = 32KB
    float*  g_sh = (float*)(smem + Bb * HDd * 2 + Bb * HDd * 4);// 32*1024 f32 = 128KB

    const int tid  = threadIdx.x;
    const int lane = tid & 31;
    const int wv   = tid >> 5;

    for (int i = tid; i < Bb * HDd; i += 256) {
        h_sh[i] = (__bf16)0.0f;
        c_sh[i] = 0.0f;
    }
    __syncthreads();

    const int hi   = lane >> 4;
    const int aRow = lane & 15;
    const int aK   = hi * 8;
    const int bK   = hi * 8;

    for (int t = 0; t < S_; ++t) {
        // ---- recurrent GEMM: g(32x1024) = h(32x256,bf16) @ W_hh^T ----
        for (int ct = 0; ct < 8; ++ct) {
            const int colBase = (wv * 8 + ct) * 16;
            const int bCol    = colBase + (lane & 15);
            const __bf16* bPtr = whh + (size_t)bCol * HDd + bK;
            v8f acc0 = {0.f, 0.f, 0.f, 0.f, 0.f, 0.f, 0.f, 0.f};
            v8f acc1 = {0.f, 0.f, 0.f, 0.f, 0.f, 0.f, 0.f, 0.f};
#pragma unroll
            for (int k0 = 0; k0 < HDd; k0 += 32) {
                v16bf bf = ld_b_frag(bPtr + k0);
                v16bf a0 = ld_a_frag(h_sh + (size_t)aRow * HDd + aK + k0);
                v16bf a1 = ld_a_frag(h_sh + (size_t)(aRow + 16) * HDd + aK + k0);
                acc0 = wmma_bf16(a0, bf, acc0);
                acc1 = wmma_bf16(a1, bf, acc1);
            }
            const int col = colBase + (lane & 15);
            const int r0  = hi * 8;
#pragma unroll
            for (int r = 0; r < 8; ++r) {
                g_sh[(size_t)(r0 + r) * NG + col]      = acc0[r];
                g_sh[(size_t)(16 + r0 + r) * NG + col] = acc1[r];
            }
        }
        __syncthreads();

        // ---- gate math + state update (masked) ----
        for (int cell = tid; cell < Bb * HDd; cell += 256) {
            const int b   = cell >> 8;          // / HDd
            const int j   = cell & (HDd - 1);
            const int len = xlen[b];
            const bool m  = (t < len);
            int tr = t;
            if (dir) { tr = len - 1 - t; if (tr < 0) tr = 0; }
            const float* gxp = gx + ((size_t)b * S_ + tr) * NG;
            float gi = g_sh[(size_t)b * NG + j]             + gxp[j];
            float gf = g_sh[(size_t)b * NG + HDd + j]       + gxp[HDd + j];
            float gg = g_sh[(size_t)b * NG + 2 * HDd + j]   + gxp[2 * HDd + j];
            float go = g_sh[(size_t)b * NG + 3 * HDd + j]   + gxp[3 * HDd + j];
            float ig = sigmoidf_fast(gi);
            float fg = sigmoidf_fast(gf);
            float cg = tanhf(gg);
            float og = sigmoidf_fast(go);
            float c_new = fg * c_sh[cell] + ig * cg;
            float h_new = og * tanhf(c_new);
            if (m) {
                c_sh[cell] = c_new;
                h_sh[cell] = (__bf16)h_new;
                out[((size_t)b * S_ + tr) * Hh + dirOff + j] = __float2bfloat16(h_new);
            }
        }
        __syncthreads();
    }
}

// ---------------- Viterbi: one wave32 per batch element, one lane per tag ----------------
__global__ __launch_bounds__(32) void viterbi_kernel(
    const float* __restrict__ em,      // (B,S,L)
    const float* __restrict__ trans,   // (L,L)
    const float* __restrict__ stt, const float* __restrict__ endt,
    const int* __restrict__ xlen,
    int* __restrict__ bp,              // (B,S-1,L)
    float* __restrict__ tags, float* __restrict__ scores, int S_) {
    __shared__ float tsh[Ll * Ll];
    const int b = blockIdx.x;
    const int j = threadIdx.x;
    for (int i = j; i < Ll * Ll; i += 32) tsh[i] = trans[i];
    __syncthreads();

    const int len = xlen[b];
    float alpha = stt[j] + em[((size_t)b * S_) * Ll + j];

    for (int t = 1; t < S_; ++t) {
        float emt  = em[((size_t)b * S_ + t) * Ll + j];
        float best = -3.4e38f;
        int   arg  = 0;
#pragma unroll 4
        for (int i = 0; i < Ll; ++i) {
            float ai = __shfl(alpha, i, 32);
            float sc = ai + tsh[i * Ll + j];
            if (sc > best) { best = sc; arg = i; }
        }
        const bool m = (t < len);
        bp[((size_t)b * (S_ - 1) + (t - 1)) * Ll + j] = m ? arg : j;
        if (m) alpha = best + emt;
    }

    float v  = alpha + endt[j];
    int  idx = j;
    for (int off = 16; off > 0; off >>= 1) {
        float v2 = __shfl_down(v, off, 32);
        int   i2 = __shfl_down(idx, off, 32);
        if (v2 > v || (v2 == v && i2 < idx)) { v = v2; idx = i2; }
    }
    if (j == 0) {
        scores[b] = v;
        int tag = idx;
        for (int t = S_ - 2; t >= 0; --t) {
            tags[(size_t)b * S_ + t + 1] = (t + 1 < len) ? (float)tag : 0.f;
            tag = bp[((size_t)b * (S_ - 1) + t) * Ll + tag];
        }
        tags[(size_t)b * S_] = (float)tag;   // t=0 always inside mask (len >= S/2)
    }
}

// ---------------- host orchestration ----------------
extern "C" void kernel_launch(void* const* d_in, const int* in_sizes, int n_in,
                              void* d_out, int out_size, void* d_ws, size_t ws_size,
                              hipStream_t stream) {
    (void)in_sizes; (void)n_in; (void)out_size; (void)ws_size;
    const int*   x    = (const int*)d_in[0];
    const int*   xlen = (const int*)d_in[1];
    const float* emb  = (const float*)d_in[2];
    const float* wih0 = (const float*)d_in[3];
    const float* whh0 = (const float*)d_in[4];
    const float* bih0 = (const float*)d_in[5];
    const float* bhh0 = (const float*)d_in[6];
    const float* wih1 = (const float*)d_in[7];
    const float* whh1 = (const float*)d_in[8];
    const float* bih1 = (const float*)d_in[9];
    const float* bhh1 = (const float*)d_in[10];
    const float* dw   = (const float*)d_in[11];
    const float* db   = (const float*)d_in[12];
    const float* trans= (const float*)d_in[13];
    const float* stt  = (const float*)d_in[14];
    const float* endt = (const float*)d_in[15];
    float* out = (float*)d_out;

    // ---- workspace partition (256B aligned) ----
    char* base = (char*)d_ws;
    size_t off = 0;
    auto take = [&](size_t bytes) -> char* {
        char* p = base + off;
        off += (bytes + 255) & ~(size_t)255;
        return p;
    };
    __hip_bfloat16* wih0_bf = (__hip_bfloat16*)take(2ull * NG * Ee  * 2);
    __hip_bfloat16* whh0_bf = (__hip_bfloat16*)take(2ull * NG * HDd * 2);
    __hip_bfloat16* wih1_bf = (__hip_bfloat16*)take(2ull * NG * Hh  * 2);
    __hip_bfloat16* whh1_bf = (__hip_bfloat16*)take(2ull * NG * HDd * 2);
    __hip_bfloat16* dw_bf   = (__hip_bfloat16*)take((size_t)Ll * Hh * 2);
    __hip_bfloat16* xemb    = (__hip_bfloat16*)take((size_t)Mrows * Ee * 2);
    __hip_bfloat16* h0      = (__hip_bfloat16*)take((size_t)Mrows * Hh * 2);
    __hip_bfloat16* h1      = (__hip_bfloat16*)take((size_t)Mrows * Hh * 2);
    float*          gx      = (float*)take(2ull * Mrows * NG * 4);
    int*            bp      = (int*)take((size_t)Bb * (Ss - 1) * Ll * 4);

    // ---- weight conversion f32 -> bf16 ----
    cvt_bf16_kernel<<<256, 256, 0, stream>>>(wih0, wih0_bf, 2 * NG * Ee);
    cvt_bf16_kernel<<<256, 256, 0, stream>>>(whh0, whh0_bf, 2 * NG * HDd);
    cvt_bf16_kernel<<<256, 256, 0, stream>>>(wih1, wih1_bf, 2 * NG * Hh);
    cvt_bf16_kernel<<<256, 256, 0, stream>>>(whh1, whh1_bf, 2 * NG * HDd);
    cvt_bf16_kernel<<<64, 256, 0, stream>>>(dw, dw_bf, Ll * Hh);

    // ---- embedding gather -> bf16 ----
    embed_kernel<<<2048, 256, 0, stream>>>(x, emb, xemb);

    // ---- zero layer outputs (masked writes only fill t < len) ----
    hipMemsetAsync(h0, 0, (size_t)Mrows * Hh * 2, stream);
    hipMemsetAsync(h1, 0, (size_t)Mrows * Hh * 2, stream);

    const int smemBytes = Bb * HDd * 2 + Bb * HDd * 4 + Bb * NG * 4; // 176KB + pad
    hipFuncSetAttribute((const void*)lstm_scan_kernel,
                        hipFuncAttributeMaxDynamicSharedMemorySize, smemBytes);

    // ---- layer 0: gx GEMM per direction, then scan ----
    {
        dim3 grid(NG / 64, Mrows / 16);
        for (int dir = 0; dir < 2; ++dir) {
            wmma_gemm_bias_kernel<<<grid, 128, 0, stream>>>(
                (const __bf16*)xemb, (const __bf16*)(wih0_bf + (size_t)dir * NG * Ee),
                bih0 + (size_t)dir * NG, bhh0 + (size_t)dir * NG,
                gx + (size_t)dir * Mrows * NG, Mrows, NG, Ee);
        }
        lstm_scan_kernel<<<2, 256, smemBytes, stream>>>(
            gx, (const __bf16*)whh0_bf, xlen, h0, Ss);
    }

    // ---- layer 1 ----
    {
        dim3 grid(NG / 64, Mrows / 16);
        for (int dir = 0; dir < 2; ++dir) {
            wmma_gemm_bias_kernel<<<grid, 128, 0, stream>>>(
                (const __bf16*)h0, (const __bf16*)(wih1_bf + (size_t)dir * NG * Hh),
                bih1 + (size_t)dir * NG, bhh1 + (size_t)dir * NG,
                gx + (size_t)dir * Mrows * NG, Mrows, NG, Hh);
        }
        lstm_scan_kernel<<<2, 256, smemBytes, stream>>>(
            gx, (const __bf16*)whh1_bf, xlen, h1, Ss);
    }

    // ---- dense projection -> emissions, straight into d_out ----
    {
        dim3 grid(1, Mrows / 16);  // N=32 -> waves 0..1 active per block
        wmma_gemm_bias_kernel<<<grid, 128, 0, stream>>>(
            (const __bf16*)h1, (const __bf16*)dw_bf, db, nullptr,
            out, Mrows, Ll, Hh);
    }

    // ---- Viterbi decode ----
    viterbi_kernel<<<Bb, 32, 0, stream>>>(
        out, trans, stt, endt, xlen, bp,
        out + (size_t)Mrows * Ll,                 // tags
        out + (size_t)Mrows * Ll + Mrows,         // scores
        Ss);
}
